// MultiHeadAttention_48713519071882
// MI455X (gfx1250) — compile-verified, matrix operand order checked by reading, not run on hardware
//
#include <hip/hip_runtime.h>

// ---------------------------------------------------------------------------
// Multi-head attention forward for MI455X (gfx1250), BF16 WMMA path.
// S=2048, B=2, D=1024, H=16, dhead=64.
// Round 3: fix async builtin signature (v4i addrspace(1)* / addrspace(3)*).
// ---------------------------------------------------------------------------

typedef __bf16 bf16_t;
typedef __attribute__((ext_vector_type(16))) __bf16 v16bf;
typedef __attribute__((ext_vector_type(8)))  float  v8f;
typedef __attribute__((ext_vector_type(4)))  int    v4i;

#define S_LEN  2048
#define BATCH  2
#define DMODEL 1024
#define NHEADS 16
#define DHEAD  64
#define MROWS  (S_LEN * BATCH)   // 4096 rows for the projection GEMMs

// Detect the gfx1250 async global->LDS builtins (toolchain dependent).
#if defined(__has_builtin)
#if __has_builtin(__builtin_amdgcn_global_load_async_to_lds_b128) && \
    __has_builtin(__builtin_amdgcn_s_wait_asynccnt)
#define USE_ASYNC_LDS 1
#endif
#endif
#ifndef USE_ASYNC_LDS
#define USE_ASYNC_LDS 0
#endif

#if USE_ASYNC_LDS
typedef __attribute__((address_space(1))) v4i* as1_v4i;
typedef __attribute__((address_space(3))) v4i* as3_v4i;
__device__ __forceinline__ void async_copy16(const bf16_t* g, bf16_t* l) {
  // global_load_async_to_lds_b128: (v4i AS1*, v4i AS3*, imm offset, imm cpol)
  __builtin_amdgcn_global_load_async_to_lds_b128(
      (as1_v4i)(v4i*)(void*)(size_t)g,   // drop const, generic -> AS1
      (as3_v4i)(v4i*)(void*)l,           // generic -> AS3
      0, 0);
}
#endif

// A/B fragment for v_wmma_f32_16x16x32_bf16: 16 bf16 per lane = two 16B loads.
union FragB16 { v16bf v; uint4 q[2]; };

__device__ __forceinline__ bf16_t f2bf(float f) {
  union { float f; unsigned u; } a; a.f = f;
  unsigned r = a.u + 0x7FFFu + ((a.u >> 16) & 1u);   // round-to-nearest-even
  union { unsigned short s; bf16_t b; } o;
  o.s = (unsigned short)(r >> 16);
  return o.b;
}

// ---------------------------------------------------------------------------
// Stage 1: f32 -> bf16 conversion (vectorized 4 elems/thread)
// ---------------------------------------------------------------------------
__global__ __launch_bounds__(256)
void mha_cvt_bf16(const float* __restrict__ in, bf16_t* __restrict__ out, int n) {
  int idx = (blockIdx.x * 256 + threadIdx.x) * 4;
  if (idx < n) {
    float4 f = *(const float4*)(in + idx);
    union { bf16_t h[4]; uint2 u; } o;
    o.h[0] = f2bf(f.x); o.h[1] = f2bf(f.y);
    o.h[2] = f2bf(f.z); o.h[3] = f2bf(f.w);
    *(uint2*)(out + idx) = o.u;
  }
}

// ---------------------------------------------------------------------------
// Stage 2: projection GEMM  out = (X @ W^T + bias) * outscale, stored bf16 in
// [B][H][S][DHEAD] layout.  One wave computes a 16(M) x 64(N) tile.
// The block's 16xDMODEL A-strip is staged once into LDS (async copy when
// available), removing the 8x redundant per-wave global reads; B = W^T
// fragments are contiguous rows of W loaded straight from global as b128s.
// ---------------------------------------------------------------------------
__global__ __launch_bounds__(256)
void mha_proj_wmma(const bf16_t* __restrict__ X,   // [MROWS][DMODEL]
                   const bf16_t* __restrict__ W,   // [DMODEL][DMODEL]
                   const float*  __restrict__ bias,// [DMODEL]
                   bf16_t* __restrict__ out,       // [B][H][S][DHEAD]
                   float outscale) {
  __shared__ __align__(16) bf16_t Atile[16][DMODEL];   // 32 KB

  const int lane = threadIdx.x & 31;
  const int wave = threadIdx.x >> 5;
  const int lh   = lane >> 4;            // lane half (0/1)
  const int l16  = lane & 15;
  const int kb   = lh * 8;               // per-half K base inside 32-chunk

  const int m0 = blockIdx.x * 16;                   // M tile origin
  const int n0 = blockIdx.y * 512 + wave * 64;      // N origin for this wave

  // ---- stage the 16 x DMODEL A-strip into LDS (2048 x 16B chunks)
#if USE_ASYNC_LDS
#pragma unroll
  for (int e = threadIdx.x; e < 2048; e += 256) {
    const int row = e >> 7;
    const int seg = (e & 127) << 3;
    async_copy16(X + (size_t)(m0 + row) * DMODEL + seg, &Atile[row][seg]);
  }
  __builtin_amdgcn_s_wait_asynccnt(0);
#else
#pragma unroll
  for (int e = threadIdx.x; e < 2048; e += 256) {
    const int row = e >> 7;
    const int seg = (e & 127) << 3;
    *(uint4*)&Atile[row][seg] =
        *(const uint4*)(X + (size_t)(m0 + row) * DMODEL + seg);
  }
#endif
  __syncthreads();

  v8f acc[4];
#pragma unroll
  for (int t = 0; t < 4; ++t)
#pragma unroll
    for (int r = 0; r < 8; ++r) acc[t][r] = 0.0f;

  for (int k0 = 0; k0 < DMODEL; k0 += 32) {
    FragB16 a;
    a.q[0] = *(const uint4*)(&Atile[l16][k0 + kb]);
    a.q[1] = *(const uint4*)(&Atile[l16][k0 + kb + 16]);
#pragma unroll
    for (int t = 0; t < 4; ++t) {
      const bf16_t* brow = W + (size_t)(n0 + t * 16 + l16) * DMODEL + k0 + kb;
      FragB16 b;
      b.q[0] = *(const uint4*)(brow);
      b.q[1] = *(const uint4*)(brow + 16);
      acc[t] = __builtin_amdgcn_wmma_f32_16x16x32_bf16(
          false, a.v, false, b.v, (short)0, acc[t], false, false);
    }
  }

  // Epilogue: C layout -> reg r holds rows r (lanes 0-15) and r+8 (lanes 16-31)
#pragma unroll
  for (int t = 0; t < 4; ++t) {
    const int dout = n0 + t * 16 + l16;
    const float bv = bias[dout];
    const int h  = dout >> 6;        // head
    const int dd = dout & 63;        // pos within head
#pragma unroll
    for (int r = 0; r < 8; ++r) {
      const int m = r + 8 * lh;
      const int n = m0 + m;          // flat row = s*BATCH + b
      const int s  = n >> 1;
      const int bb = n & 1;
      const float val = (acc[t][r] + bv) * outscale;
      out[(((size_t)bb * NHEADS + h) * S_LEN + s) * DHEAD + dd] = f2bf(val);
    }
  }
}

// ---------------------------------------------------------------------------
// Stage 3: flash attention.  Block = 256 thr = 8 waves; each wave owns 16 query
// rows; block covers 128 rows of one (b,h).  Streams keys/values in j-steps
// of 32 with online softmax.  K tile staged row-major into LDS (async copy
// when available, shared by all 8 waves); V tile staged transposed for the
// P@V B-fragments.  Q pre-scaled by 1/sqrt(d) in the projection.
// ---------------------------------------------------------------------------
__global__ __launch_bounds__(256)
void mha_attn_wmma(const bf16_t* __restrict__ Q,   // [B][H][S][64] (pre-scaled)
                   const bf16_t* __restrict__ K,   // [B][H][S][64]
                   const bf16_t* __restrict__ V,   // [B][H][S][64]
                   float* __restrict__ out) {      // [S][B][DMODEL]
  __shared__ __align__(16) bf16_t Kt[32][DHEAD];    // K tile row-major (4 KB)
  __shared__ __align__(16) bf16_t Vt[DHEAD][32];    // V tile transposed (4 KB)
  __shared__ __align__(16) bf16_t Pl[8][16][32];    // per-wave P tile (8 KB)

  const int lane = threadIdx.x & 31;
  const int wave = threadIdx.x >> 5;
  const int lh   = lane >> 4;
  const int l16  = lane & 15;
  const int kb   = lh * 8;

  const int bh = blockIdx.y;            // b*NHEADS + h
  const int bb = bh >> 4;
  const int h  = bh & 15;
  const int i0 = blockIdx.x * 128 + wave * 16;

  const bf16_t* Qb = Q + (size_t)bh * S_LEN * DHEAD;
  const bf16_t* Kb = K + (size_t)bh * S_LEN * DHEAD;
  const bf16_t* Vb = V + (size_t)bh * S_LEN * DHEAD;

  // Per-thread staging coordinates: 32 rows x 64 cols, one b128 per thread.
  const int srow = threadIdx.x >> 3;        // 0..31
  const int sseg = (threadIdx.x & 7) << 3;  // 0,8,..,56

  // Load the wave's Q fragments once (two K-splits of 32 over d=64).
  FragB16 qf[2];
  {
    const bf16_t* qrow = Qb + (size_t)(i0 + l16) * DHEAD;
    qf[0].q[0] = *(const uint4*)(qrow + kb);
    qf[0].q[1] = *(const uint4*)(qrow + kb + 16);
    qf[1].q[0] = *(const uint4*)(qrow + 32 + kb);
    qf[1].q[1] = *(const uint4*)(qrow + 32 + kb + 16);
  }

  v8f o[4];
#pragma unroll
  for (int t = 0; t < 4; ++t)
#pragma unroll
    for (int r = 0; r < 8; ++r) o[t][r] = 0.0f;

  float mrow[8], lrow[8];
#pragma unroll
  for (int r = 0; r < 8; ++r) { mrow[r] = -3.0e38f; lrow[r] = 0.0f; }

  for (int j0 = 0; j0 < S_LEN; j0 += 32) {
    __syncthreads();   // previous-iteration LDS reads done before overwrite

    // ---- stage K[j0:j0+32][0:64] row-major into LDS
#if USE_ASYNC_LDS
    async_copy16(Kb + (size_t)(j0 + srow) * DHEAD + sseg, &Kt[srow][sseg]);
#else
    *(uint4*)&Kt[srow][sseg] =
        *(const uint4*)(Kb + (size_t)(j0 + srow) * DHEAD + sseg);
#endif

    // ---- stage V[j0:j0+32][0:64] transposed into LDS (b128 read + scatter)
    {
      union { uint4 q; bf16_t e[8]; } vv;
      vv.q = *(const uint4*)(Vb + (size_t)(j0 + srow) * DHEAD + sseg);
#pragma unroll
      for (int x = 0; x < 8; ++x) Vt[sseg + x][srow] = vv.e[x];
    }

    // ---- prefetch next j-tile of K and V while this one is consumed
    if (j0 + 32 < S_LEN) {
      __builtin_prefetch(Kb + (size_t)(j0 + 32 + srow) * DHEAD + sseg, 0, 1);
      __builtin_prefetch(Vb + (size_t)(j0 + 32 + srow) * DHEAD + sseg, 0, 1);
    }

#if USE_ASYNC_LDS
    __builtin_amdgcn_s_wait_asynccnt(0);
#endif
    __syncthreads();

    // ---- S = Q K^T for two 16-wide j-subtiles (fragments from LDS)
    v8f sa[2];
#pragma unroll
    for (int jt = 0; jt < 2; ++jt) {
      v8f s;
#pragma unroll
      for (int r = 0; r < 8; ++r) s[r] = 0.0f;
#pragma unroll
      for (int ks = 0; ks < 2; ++ks) {
        FragB16 kf;
        kf.q[0] = *(const uint4*)(&Kt[jt * 16 + l16][ks * 32 + kb]);
        kf.q[1] = *(const uint4*)(&Kt[jt * 16 + l16][ks * 32 + kb + 16]);
        s = __builtin_amdgcn_wmma_f32_16x16x32_bf16(
            false, qf[ks].v, false, kf.v, (short)0, s, false, false);
      }
      sa[jt] = s;
    }

    // ---- online softmax over the 32 new columns.
    // Reg r holds row r (lanes 0-15) / row r+8 (lanes 16-31); xor masks <=8
    // keep the butterfly inside each 16-lane half, i.e. per-row reductions.
#pragma unroll
    for (int r = 0; r < 8; ++r) {
      float cmax = fmaxf(sa[0][r], sa[1][r]);
#pragma unroll
      for (int msk = 8; msk >= 1; msk >>= 1)
        cmax = fmaxf(cmax, __shfl_xor(cmax, msk, 32));
      const float mnew  = fmaxf(mrow[r], cmax);
      const float alpha = __expf(mrow[r] - mnew);
      const float p0 = __expf(sa[0][r] - mnew);
      const float p1 = __expf(sa[1][r] - mnew);
      float rs = p0 + p1;
#pragma unroll
      for (int msk = 8; msk >= 1; msk >>= 1)
        rs += __shfl_xor(rs, msk, 32);
      lrow[r] = lrow[r] * alpha + rs;
      mrow[r] = mnew;
      sa[0][r] = p0;
      sa[1][r] = p1;
#pragma unroll
      for (int t = 0; t < 4; ++t) o[t][r] *= alpha;
    }

    // ---- re-layout P (C-format) -> A-format through wave-private LDS
#pragma unroll
    for (int r = 0; r < 8; ++r) {
      const int row = r + 8 * lh;
      Pl[wave][row][l16]      = f2bf(sa[0][r]);
      Pl[wave][row][l16 + 16] = f2bf(sa[1][r]);
    }
    __syncthreads();   // order LDS stores vs. cross-lane loads

    FragB16 pf;
    pf.q[0] = *(const uint4*)(&Pl[wave][l16][kb]);
    pf.q[1] = *(const uint4*)(&Pl[wave][l16][kb + 16]);

    // ---- O += P @ V  (K dim = 32, exactly one WMMA per 16-wide N tile)
#pragma unroll
    for (int t = 0; t < 4; ++t) {
      FragB16 vf;
      vf.q[0] = *(const uint4*)(&Vt[t * 16 + l16][kb]);
      vf.q[1] = *(const uint4*)(&Vt[t * 16 + l16][kb + 16]);
      o[t] = __builtin_amdgcn_wmma_f32_16x16x32_bf16(
          false, pf.v, false, vf.v, (short)0, o[t], false, false);
    }
  }

  // ---- epilogue: normalize and store f32 to [S][B][DMODEL]
#pragma unroll
  for (int r = 0; r < 8; ++r) {
    const int row = r + 8 * lh;
    const int i = i0 + row;
    const float inv = 1.0f / lrow[r];
#pragma unroll
    for (int t = 0; t < 4; ++t) {
      const int dd = t * 16 + l16;
      out[((size_t)i * BATCH + bb) * DMODEL + h * DHEAD + dd] = o[t][r] * inv;
    }
  }
}

// ---------------------------------------------------------------------------
// Launch
// ---------------------------------------------------------------------------
extern "C" void kernel_launch(void* const* d_in, const int* in_sizes, int n_in,
                              void* d_out, int out_size, void* d_ws, size_t ws_size,
                              hipStream_t stream) {
  const float* q_in = (const float*)d_in[0];
  const float* k_in = (const float*)d_in[1];
  const float* v_in = (const float*)d_in[2];
  const float* Wq   = (const float*)d_in[3];
  const float* bq   = (const float*)d_in[4];
  const float* Wk   = (const float*)d_in[5];
  const float* bk   = (const float*)d_in[6];
  const float* Wv   = (const float*)d_in[7];
  const float* bv   = (const float*)d_in[8];
  float* out = (float*)d_out;

  const size_t Xe = (size_t)MROWS * DMODEL;    // 4,194,304 elems
  const size_t We = (size_t)DMODEL * DMODEL;   // 1,048,576 elems

  bf16_t* p  = (bf16_t*)d_ws;
  bf16_t* xq = p; p += Xe;
  bf16_t* xk = p; p += Xe;
  bf16_t* xv = p; p += Xe;
  bf16_t* wq = p; p += We;
  bf16_t* wk = p; p += We;
  bf16_t* wv = p; p += We;
  bf16_t* Pq = p; p += Xe;   // [B][H][S][64], pre-scaled by 1/sqrt(d)
  bf16_t* Pk = p; p += Xe;
  bf16_t* Pv = p; p += Xe;

  dim3 blk(256);

  // Stage 1: convert activations + weights to bf16
  mha_cvt_bf16<<<(unsigned)(Xe / 1024), blk, 0, stream>>>(q_in, xq, (int)Xe);
  mha_cvt_bf16<<<(unsigned)(Xe / 1024), blk, 0, stream>>>(k_in, xk, (int)Xe);
  mha_cvt_bf16<<<(unsigned)(Xe / 1024), blk, 0, stream>>>(v_in, xv, (int)Xe);
  mha_cvt_bf16<<<(unsigned)(We / 1024), blk, 0, stream>>>(Wq, wq, (int)We);
  mha_cvt_bf16<<<(unsigned)(We / 1024), blk, 0, stream>>>(Wk, wk, (int)We);
  mha_cvt_bf16<<<(unsigned)(We / 1024), blk, 0, stream>>>(Wv, wv, (int)We);

  // Stage 2: projections (1/sqrt(64) = 0.125 folded into Q)
  dim3 pgrid(MROWS / 16, DMODEL / 512);
  mha_proj_wmma<<<pgrid, blk, 0, stream>>>(xq, wq, bq, Pq, 0.125f);
  mha_proj_wmma<<<pgrid, blk, 0, stream>>>(xk, wk, bk, Pk, 1.0f);
  mha_proj_wmma<<<pgrid, blk, 0, stream>>>(xv, wv, bv, Pv, 1.0f);

  // Stage 3: flash attention
  dim3 agrid(S_LEN / 128, BATCH * NHEADS);
  mha_attn_wmma<<<agrid, blk, 0, stream>>>(Pq, Pk, Pv, out);
}
